// GCN_13657996001673
// MI455X (gfx1250) — matrix operand order repaired and checked by reading, not compiled, not run on hardware
//
#include <hip/hip_runtime.h>
#include <hip/hip_bf16.h>

typedef __attribute__((ext_vector_type(16))) __bf16 v16bf;
typedef __attribute__((ext_vector_type(8)))  __bf16 v8bf;
typedef __attribute__((ext_vector_type(8)))  float  v8f;

#define DF 128      // feature dim
#define KCB 16      // codebook size
#define NQ 3        // quantizer stages
#define NL 3        // layers
#define COMMIT_W 0.25f
#define WT_STRIDE 136   // padded bf16 row stride (272 B = 68 dwords, bank-friendly)
#define RES_STRIDE 132  // padded f32 row stride (528 B = 132 dwords, bank-friendly)

__device__ __forceinline__ __bf16 tobf(float f) { return (__bf16)f; }

// 16-byte async copy global -> LDS (CDNA5 GLOBAL_LOAD_ASYNC_TO_LDS_B128, ASYNCcnt)
__device__ __forceinline__ void async_ld_b128(void* lds_dst, const void* gsrc) {
    unsigned l = (unsigned)(unsigned long long)lds_dst;           // LDS byte offset
    unsigned long long g = (unsigned long long)gsrc;
    asm volatile("global_load_async_to_lds_b128 %0, %1, off"
                 :: "v"(l), "v"(g) : "memory");
}
__device__ __forceinline__ void wait_async0() {
    asm volatile("s_wait_asynccnt 0x0" ::: "memory");
}

// ---------------- init: zero degree accumulator and loss ----------------
__global__ void k_init(float* deg, float* loss, int n) {
    int i = blockIdx.x * blockDim.x + threadIdx.x;
    if (i < n) deg[i] = 0.0f;
    if (i == 0) *loss = 0.0f;
}

// ------- L2-normalize codebook rows; emit fp32 copy + bf16 (B-layout) copy -------
__global__ void k_cbnorm(const float* __restrict__ cb, float* __restrict__ cbn,
                         __bf16* __restrict__ cbnbf, int rows) {
    int r = blockIdx.x * blockDim.x + threadIdx.x;
    if (r >= rows) return;
    const float* src = cb + (size_t)r * DF;
    float ss = 0.0f;
    for (int k = 0; k < DF; ++k) ss += src[k] * src[k];
    float inv = 1.0f / (sqrtf(ss) + 1e-12f);
    float* dst = cbn + (size_t)r * DF;
    __bf16* dbf = cbnbf + (size_t)r * DF;
    for (int k = 0; k < DF; ++k) {
        float v = src[k] * inv;
        dst[k] = v;
        dbf[k] = tobf(v);
    }
}

// ---------------- degree via hardware float atomics ----------------
__global__ void k_deg(const int* __restrict__ erow, float* deg, int e) {
    int i = blockIdx.x * blockDim.x + threadIdx.x;
    if (i < e) unsafeAtomicAdd(&deg[erow[i]], 1.0f);
}

// ---------------- norm = rsqrt(1 + deg), in place ----------------
__global__ void k_norm(float* normv, int n) {
    int i = blockIdx.x * blockDim.x + threadIdx.x;
    if (i < n) normv[i] = rsqrtf(1.0f + normv[i]);
}

// ---------------- h = x @ W + b  (bf16 WMMA, fp32 accumulate) ----------------
// Block: 256 thr = 8 waves; block covers 16 rows, wave w does columns [16w,16w+16).
// W transposed into LDS as bf16 with padded stride -> B frag = 2x ds_load_b128.
// A rows clamped (not branched) so loads pipeline; stores are guarded.
__global__ __launch_bounds__(256) void k_gemm(const float* __restrict__ x,
                                              const float* __restrict__ w,
                                              const float* __restrict__ b,
                                              float* __restrict__ h, int n) {
    __shared__ __bf16 Wt[DF * WT_STRIDE];           // [col][k], ~34 KB
    int tid = threadIdx.x;
    for (int i = tid; i < DF * DF; i += 256) {
        int k = i >> 7, nc = i & 127;               // coalesced global read
        Wt[nc * WT_STRIDE + k] = tobf(w[i]);
    }
    __syncthreads();

    int wv = tid >> 5, lane = tid & 31, half = lane >> 4, m = lane & 15;
    int row  = blockIdx.x * 16 + m;
    int rowc = row < n ? row : (n - 1);             // clamp: pad rows compute garbage,
    int colg = wv * 16 + m;                         // but their stores are dropped

    const float4* xr4 = (const float4*)(x + (size_t)rowc * DF);
    const __bf16* wrow = Wt + colg * WT_STRIDE;

    v8f c = {0.f,0.f,0.f,0.f,0.f,0.f,0.f,0.f};
    #pragma unroll
    for (int kb = 0; kb < 4; ++kb) {                // K = 128 in 4 chunks of 32
        // A 16x32 layout: lane half selects K ranges {h*8..h*8+7, 16+h*8..}
        int f0 = kb * 8 + half * 2;                 // float4 index of K = kb*32+half*8
        float4 p0 = xr4[f0], p1 = xr4[f0 + 1];
        float4 p2 = xr4[f0 + 4], p3 = xr4[f0 + 5];
        v16bf a;
        a[0]=tobf(p0.x); a[1]=tobf(p0.y); a[2]=tobf(p0.z); a[3]=tobf(p0.w);
        a[4]=tobf(p1.x); a[5]=tobf(p1.y); a[6]=tobf(p1.z); a[7]=tobf(p1.w);
        a[8]=tobf(p2.x); a[9]=tobf(p2.y); a[10]=tobf(p2.z); a[11]=tobf(p2.w);
        a[12]=tobf(p3.x); a[13]=tobf(p3.y); a[14]=tobf(p3.z); a[15]=tobf(p3.w);
        // B 32x16 layout: lane = column, K ranges {h*8..h*8+7, 16+h*8..} per half
        const v8bf* bp = (const v8bf*)(wrow + kb * 32 + half * 8);
        v8bf b0 = bp[0], b1 = bp[2];
        v16bf bb = __builtin_shufflevector(b0, b1, 0,1,2,3,4,5,6,7,8,9,10,11,12,13,14,15);
        c = __builtin_amdgcn_wmma_f32_16x16x32_bf16(false, a, false, bb,
                                                    (short)0, c, false, false);
    }
    float bias = b[colg];
    #pragma unroll
    for (int r = 0; r < 8; ++r) {                   // C: VGPR r -> row r + 8*half
        int orow = blockIdx.x * 16 + r + half * 8;
        if (orow < n) h[(size_t)orow * DF + colg] = c[r] + bias;
    }
}

// ---------------- per layer: hn = norm*h, agg = 0 (one fused pass) ----------------
__global__ void k_prep(const float* __restrict__ h, const float* __restrict__ normv,
                       float* __restrict__ hn, float* __restrict__ agg, long long total4) {
    long long i = (long long)blockIdx.x * blockDim.x + threadIdx.x;
    long long stride = (long long)gridDim.x * blockDim.x;
    const float4* h4 = (const float4*)h;
    float4* hn4 = (float4*)hn;
    float4* ag4 = (float4*)agg;
    for (; i < total4; i += stride) {
        float nv = normv[i >> 5];
        float4 hv = h4[i];
        hn4[i] = make_float4(hv.x * nv, hv.y * nv, hv.z * nv, hv.w * nv);
        ag4[i] = make_float4(0.f, 0.f, 0.f, 0.f);
    }
}

// ---------------- SpMM scatter: agg[row] += hn[col], one wave per edge ----------------
__global__ __launch_bounds__(256) void k_scatter(const int* __restrict__ erow,
                                                 const int* __restrict__ ecol,
                                                 const float* __restrict__ hn,
                                                 float* __restrict__ agg, int e) {
    int g = (blockIdx.x * blockDim.x + threadIdx.x) >> 5;
    int lane = threadIdx.x & 31;
    if (g >= e) return;
    int r = erow[g], c = ecol[g];
    float4 v = ((const float4*)(hn + (size_t)c * DF))[lane];
    float* dst = agg + (size_t)r * DF + lane * 4;
    unsafeAtomicAdd(dst + 0, v.x);
    unsafeAtomicAdd(dst + 1, v.y);
    unsafeAtomicAdd(dst + 2, v.z);
    unsafeAtomicAdd(dst + 3, v.w);
}

// ---- fused: x1 = norm*(agg+hn); h += x1; residual-VQ (3 stages, WMMA sim) ----
// Block: 128 thr = 4 waves; wave handles 16 nodes. Residual tile in padded LDS,
// bf16 codebooks staged via async-to-LDS; qv subtraction reads fp32 from global.
__global__ __launch_bounds__(128) void k_vq(const float* __restrict__ hn,
                                            const float* __restrict__ agg,
                                            const float* __restrict__ normv,
                                            const float* __restrict__ cbn_l,
                                            const __bf16* __restrict__ cbb_l,
                                            float* __restrict__ h,
                                            float* __restrict__ loss,
                                            float* __restrict__ ids,
                                            int n, int layer) {
    __shared__ __bf16 cbb[NQ * KCB * WT_STRIDE];    // ~12.75 KB, B-fragment-ready
    __shared__ float  res[4 * 16 * RES_STRIDE];     // ~33 KB residual tiles
    __shared__ int    idxs[4 * 16];

    int tid = threadIdx.x;
    // stage bf16 codebooks with CDNA5 async-to-LDS (16 B per lane-op, 6 iters)
    for (int c = tid; c < NQ * KCB * (DF / 8); c += 128) {
        int r = c >> 4, off = (c & 15) * 8;
        async_ld_b128(cbb + r * WT_STRIDE + off, cbb_l + r * DF + off);
    }

    int wv = tid >> 5, lane = tid & 31, half = lane >> 4, nd = lane & 15;
    int nb = blockIdx.x * 64 + wv * 16;
    int row = nb + nd;
    float* myres = res + (size_t)(wv * 16 + nd) * RES_STRIDE;
    float4* r4 = (float4*)myres;

    // x1 = norm*(agg+hn) -> LDS residual; h += x1 (overlaps the async copy)
    if (row < n) {
        float nv = normv[row];
        const float4* a4 = (const float4*)(agg + (size_t)row * DF);
        const float4* s4 = (const float4*)(hn + (size_t)row * DF);
        float4* h4 = (float4*)(h + (size_t)row * DF);
        #pragma unroll
        for (int j = 0; j < 16; ++j) {
            int k4 = half * 16 + j;
            float4 av = a4[k4], sv = s4[k4];
            float4 x1 = make_float4(nv * (av.x + sv.x), nv * (av.y + sv.y),
                                    nv * (av.z + sv.z), nv * (av.w + sv.w));
            r4[k4] = x1;
            float4 ho = h4[k4];
            h4[k4] = make_float4(ho.x + x1.x, ho.y + x1.y, ho.z + x1.z, ho.w + x1.w);
        }
    } else {
        #pragma unroll
        for (int j = 0; j < 16; ++j)
            r4[half * 16 + j] = make_float4(0.f, 0.f, 0.f, 0.f);
    }
    wait_async0();                                   // ASYNCcnt == 0: cbb resident
    __syncthreads();

    const float lscale = COMMIT_W / ((float)n * (float)DF);

    for (int q = 0; q < NQ; ++q) {
        // inverse L2 norm of this lane's node (halves paired via shfl_xor 16)
        float ss = 0.0f;
        #pragma unroll
        for (int j = 0; j < 16; ++j) {
            float4 v = r4[half * 16 + j];
            ss += v.x * v.x + v.y * v.y + v.z * v.z + v.w * v.w;
        }
        ss += __shfl_xor(ss, 16, 32);
        float inv = 1.0f / (sqrtf(ss) + 1e-12f);

        // sim tile [16 nodes x 16 codes] = l2norm(res) @ cb^T
        const __bf16* brow = cbb + (q * KCB + nd) * WT_STRIDE;
        v8f c = {0.f,0.f,0.f,0.f,0.f,0.f,0.f,0.f};
        #pragma unroll
        for (int kb = 0; kb < 4; ++kb) {
            int f0 = kb * 8 + half * 2;
            float4 p0 = r4[f0], p1 = r4[f0 + 1];
            float4 p2 = r4[f0 + 4], p3 = r4[f0 + 5];
            v16bf a;
            a[0]=tobf(p0.x*inv); a[1]=tobf(p0.y*inv); a[2]=tobf(p0.z*inv); a[3]=tobf(p0.w*inv);
            a[4]=tobf(p1.x*inv); a[5]=tobf(p1.y*inv); a[6]=tobf(p1.z*inv); a[7]=tobf(p1.w*inv);
            a[8]=tobf(p2.x*inv); a[9]=tobf(p2.y*inv); a[10]=tobf(p2.z*inv); a[11]=tobf(p2.w*inv);
            a[12]=tobf(p3.x*inv); a[13]=tobf(p3.y*inv); a[14]=tobf(p3.z*inv); a[15]=tobf(p3.w*inv);
            const v8bf* bp = (const v8bf*)(brow + kb * 32 + half * 8);
            v8bf b0 = bp[0], b1 = bp[2];
            v16bf bb = __builtin_shufflevector(b0, b1, 0,1,2,3,4,5,6,7,8,9,10,11,12,13,14,15);
            c = __builtin_amdgcn_wmma_f32_16x16x32_bf16(false, a, false, bb,
                                                        (short)0, c, false, false);
        }

        // per-node argmax across the 16 lanes of each half (wave32 shuffle tree)
        #pragma unroll
        for (int r = 0; r < 8; ++r) {
            float v = c[r];
            int bi = nd;
            #pragma unroll
            for (int off = 1; off < 16; off <<= 1) {
                float ov = __shfl_xor(v, off, 32);
                int   oi = __shfl_xor(bi, off, 32);
                if (ov > v || (ov == v && oi < bi)) { v = ov; bi = oi; }
            }
            int node = r + half * 8;                 // C layout: VGPR r -> row r+8*half
            if (nd == 0) {
                idxs[wv * 16 + node] = bi;
                int grow = nb + node;
                if (grow < n)
                    ids[(size_t)grow * (NL * NQ) + layer * NQ + q] = (float)bi;
            }
        }
        __syncthreads();

        // residual -= cbn[idx] (fp32 from global/L2); loss += mean(residual_new^2)
        float sq = 0.0f;
        if (row < n) {
            const float4* q4 = (const float4*)(cbn_l + (size_t)(q * KCB + idxs[wv * 16 + nd]) * DF);
            #pragma unroll
            for (int j = 0; j < 16; ++j) {
                int k4 = half * 16 + j;
                float4 rv = r4[k4], qq = q4[k4];
                float4 nr = make_float4(rv.x - qq.x, rv.y - qq.y, rv.z - qq.z, rv.w - qq.w);
                r4[k4] = nr;
                sq += nr.x * nr.x + nr.y * nr.y + nr.z * nr.z + nr.w * nr.w;
            }
        }
        #pragma unroll
        for (int off = 16; off > 0; off >>= 1) sq += __shfl_down(sq, off, 32);
        if (lane == 0) unsafeAtomicAdd(loss, sq * lscale);
        __syncthreads();
    }
}

extern "C" void kernel_launch(void* const* d_in, const int* in_sizes, int n_in,
                              void* d_out, int out_size, void* d_ws, size_t ws_size,
                              hipStream_t stream) {
    const float* x    = (const float*)d_in[0];
    const float* w_in = (const float*)d_in[1];
    const float* b_in = (const float*)d_in[2];
    const float* cbs  = (const float*)d_in[3];
    const int*   erow = (const int*)d_in[4];
    const int*   ecol = (const int*)d_in[5];

    int n = in_sizes[0] / DF;
    int e = in_sizes[4];
    size_t nd = (size_t)n * DF;

    // outputs: h [N*D], loss [1], ids [N*9] (flat concat, fp32)
    float* outh = (float*)d_out;
    float* loss = outh + nd;
    float* ids  = loss + 1;

    // workspace: norm [N], hn [N*D], agg [N*D], cbn fp32 [144*128], cbn bf16 [144*128]
    float* wsf   = (float*)d_ws;
    float* normv = wsf;
    float* hnb   = wsf + n;
    float* aggb  = hnb + nd;
    float* cbn   = aggb + nd;
    __bf16* cbnbf = (__bf16*)(cbn + (size_t)NL * NQ * KCB * DF);

    k_init<<<(n + 255) / 256, 256, 0, stream>>>(normv, loss, n);
    k_cbnorm<<<(NL * NQ * KCB + 127) / 128, 128, 0, stream>>>(cbs, cbn, cbnbf, NL * NQ * KCB);
    k_deg<<<(e + 255) / 256, 256, 0, stream>>>(erow, normv, e);
    k_norm<<<(n + 255) / 256, 256, 0, stream>>>(normv, n);
    k_gemm<<<(n + 15) / 16, 256, 0, stream>>>(x, w_in, b_in, outh, n);

    long long total4 = (long long)nd / 4;
    for (int L = 0; L < NL; ++L) {
        k_prep<<<2048, 256, 0, stream>>>(outh, normv, hnb, aggb, total4);
        k_scatter<<<(e + 7) / 8, 256, 0, stream>>>(erow, ecol, hnb, aggb, e);
        k_vq<<<(n + 63) / 64, 128, 0, stream>>>(hnb, aggb, normv,
                                                cbn + (size_t)L * NQ * KCB * DF,
                                                cbnbf + (size_t)L * NQ * KCB * DF,
                                                outh, loss, ids, n, L);
    }
}
